// MSDeformAttn2dBase_12506944765971
// MI455X (gfx1250) — matrix operand order repaired and checked by reading, not compiled, not run on hardware
//
#include <hip/hip_runtime.h>

#define B_  4
#define Q_  4096
#define C_  256
#define M_  8
#define L_  4
#define P_  4
#define D_  32
#define NV_ 21760
#define BQ_ (B_*Q_)
#define BN_ (B_*NV_)

typedef __bf16 bf16_t;
typedef bf16_t v16bf __attribute__((ext_vector_type(16)));
typedef bf16_t v2bf  __attribute__((ext_vector_type(2)));
typedef float  v8f   __attribute__((ext_vector_type(8)));

union BfVec { v16bf v; uint4 u[2]; };

__device__ __forceinline__ bf16_t f2bf(float x) {
    // round-to-nearest-even f32 -> bf16 via integer math (fallback path)
    unsigned u = __float_as_uint(x);
    unsigned r = (u + 0x7FFFu + ((u >> 16) & 1u)) >> 16;
    unsigned short s = (unsigned short)r;
    bf16_t b;
    __builtin_memcpy(&b, &s, 2);
    return b;
}

// pack two f32 into adjacent bf16 elements of a v16bf fragment
__device__ __forceinline__ void set2bf(v16bf& a, int i2, float lo, float hi) {
#if __has_builtin(__builtin_amdgcn_cvt_pk_bf16_f32)
    v2bf pk = __builtin_amdgcn_cvt_pk_bf16_f32(lo, hi);
    a[2 * i2]     = pk[0];
    a[2 * i2 + 1] = pk[1];
#else
    a[2 * i2]     = f2bf(lo);
    a[2 * i2 + 1] = f2bf(hi);
#endif
}

__device__ __forceinline__ float lane_bcast_f(float v, int srcLane) {
    return __int_as_float(__builtin_amdgcn_ds_bpermute(srcLane << 2, __float_as_int(v)));
}
__device__ __forceinline__ float xor_shfl_f(float v, int mask) {
    int lane = (int)(threadIdx.x & 31u);
    return lane_bcast_f(v, lane ^ mask);
}

// ---------------------------------------------------------------------------
// Kernel 0: convert + transpose a weight matrix W[K][N] (f32, row-major) into
// Wt[N][K] (bf16) so WMMA B-fragments read two runs of 8 contiguous bf16 per lane.
// ---------------------------------------------------------------------------
__global__ void convert_wT_kernel(const float* __restrict__ src, bf16_t* __restrict__ dst,
                                  int K, int N) {
    int t = blockIdx.x * blockDim.x + threadIdx.x;
    if (t >= K * N) return;
    int k = t % K;
    int n = t / K;
    dst[(size_t)n * K + k] = f2bf(src[(size_t)k * N + n]);
}

// ---------------------------------------------------------------------------
// Kernel 1: GEMM  C[rows][N] = A[rows][K](f32) x Bt[N][K](bf16) + bias
// One wave computes a 16-row x 64-col strip (4 accumulators), K-step 32,
// via v_wmma_f32_16x16x32_bf16. rows % 16 == 0, K % 32 == 0, N % 64 == 0.
// STORE_MODE 0: row-major [rows][N]
// STORE_MODE 1: value-proj layout: row=(b,n), col=(m,d) -> out[((b*8+m)*NV_+n)*32+d]
//               (NV_ is compile-time -> division becomes mul/shift)
// ---------------------------------------------------------------------------
template <int STORE_MODE>
__global__ void gemm_wmma_bf16_kernel(const float* __restrict__ A,
                                      const bf16_t* __restrict__ Bt,
                                      const float* __restrict__ bias,
                                      float* __restrict__ Cout,
                                      int rows, int K, int N) {
    const int lane = (int)(threadIdx.x & 31u);
    const int wib  = (int)(threadIdx.x >> 5u);
    const int gwid = blockIdx.x * (blockDim.x >> 5u) + wib;
    const int colGroups = N >> 6;
    const int rowTiles  = rows >> 4;
    if (gwid >= rowTiles * colGroups) return;      // wave-uniform: EXEC stays all-ones
    const int rowTile = gwid / colGroups;
    const int cg      = gwid - rowTile * colGroups;

    const int aRow = rowTile * 16 + (lane & 15);
    const int kA   = (lane >> 4) ? 8 : 0;          // A 16x32 bf16 lane layout
    const int bCol = lane & 15;
    const int kB   = (lane >> 4) ? 8 : 0;          // B 32x16 bf16 lane layout

    v8f acc[4] = {};

    for (int k0 = 0; k0 < K; k0 += 32) {
        const float* ap = A + (size_t)aRow * K + k0 + kA;
        float4 f0 = *(const float4*)(ap);
        float4 f1 = *(const float4*)(ap + 4);
        float4 f2 = *(const float4*)(ap + 16);
        float4 f3 = *(const float4*)(ap + 20);
        v16bf a;
        set2bf(a, 0, f0.x, f0.y);
        set2bf(a, 1, f0.z, f0.w);
        set2bf(a, 2, f1.x, f1.y);
        set2bf(a, 3, f1.z, f1.w);
        set2bf(a, 4, f2.x, f2.y);
        set2bf(a, 5, f2.z, f2.w);
        set2bf(a, 6, f3.x, f3.y);
        set2bf(a, 7, f3.z, f3.w);

#pragma unroll
        for (int t4 = 0; t4 < 4; ++t4) {
            const bf16_t* bp = Bt + (size_t)(cg * 64 + t4 * 16 + bCol) * K + k0 + kB;
            BfVec bv;
            bv.u[0] = *(const uint4*)(bp);
            bv.u[1] = *(const uint4*)(bp + 16);
            acc[t4] = __builtin_amdgcn_wmma_f32_16x16x32_bf16(
                false, a, false, bv.v, (short)0, acc[t4], false, false);
        }
    }

    // D layout: lane holds col = (lane&15), rows M = r + 8*(lane>>4) in VGPR r
    const int dRowBase = rowTile * 16 + ((lane >> 4) ? 8 : 0);
    const int dCol     = lane & 15;
#pragma unroll
    for (int t4 = 0; t4 < 4; ++t4) {
        const int col = cg * 64 + t4 * 16 + dCol;
        const float bv = bias[col];
#pragma unroll
        for (int r = 0; r < 8; ++r) {
            const int row = dRowBase + r;
            const float val = acc[t4][r] + bv;
            if (STORE_MODE == 0) {
                Cout[(size_t)row * N + col] = val;
            } else {
                const int bb = row / NV_;          // compile-time const -> mul/shift
                const int n  = row - bb * NV_;
                const int m  = col >> 5;
                const int d  = col & 31;
                Cout[(((size_t)(bb * 8 + m) * (size_t)NV_) + n) * 32 + d] = val;
            }
        }
    }
}

// ---------------------------------------------------------------------------
// Kernel 2: deformable sampling. One wave32 per (b,q,m); lane = d in [0,32).
// Lanes 0..15 own point j = l*4+p: softmax weight + sampling coords, broadcast
// via ds_bpermute. Each corner fetch is wave-uniform -> coalesced 128B load.
// Output t[b*Q+q][m*32+d] = sum_j attn_j * bilinear(v_p[b][m], loc_j)[d]
// ---------------------------------------------------------------------------
__global__ void msda_sample_kernel(const float* __restrict__ attn_raw,   // [BQ][128]
                                   const float* __restrict__ off_raw,    // [BQ][256]
                                   const float* __restrict__ pref,       // [B][Q][L][2]
                                   const float* __restrict__ v_p,        // [B][M][NV][32]
                                   const int*   __restrict__ shapes,     // [L][2] (H,W)
                                   const int*   __restrict__ level_index,// [L]
                                   float* __restrict__ tOut) {           // [BQ][256]
    const int lane = (int)(threadIdx.x & 31u);
    const int gwid = blockIdx.x * (blockDim.x >> 5u) + (int)(threadIdx.x >> 5u);
    if (gwid >= B_ * Q_ * M_) return;              // wave-uniform
    const int m  = gwid & (M_ - 1);
    const int bq = gwid >> 3;
    const int b  = bq >> 12;                        // Q_ == 4096

    // ---- per-point setup (point j on lane j, duplicated on lane j+16) ----
    const int j  = lane & 15;
    const int lj = j >> 2;
    const float aval = attn_raw[(size_t)bq * 128 + m * 16 + j];

    float mx = aval;
#pragma unroll
    for (int s = 16; s >= 1; s >>= 1) mx = fmaxf(mx, xor_shfl_f(mx, s));
    float e = __expf(aval - mx);
    float ssum = e;
#pragma unroll
    for (int s = 16; s >= 1; s >>= 1) ssum += xor_shfl_f(ssum, s);
    const float w = e * (2.0f / ssum);              // 32-lane sum double-counts 16 values

    const float ox = off_raw[(size_t)bq * 256 + m * 32 + 2 * j];
    const float oy = off_raw[(size_t)bq * 256 + m * 32 + 2 * j + 1];
    const int   Hl = shapes[lj * 2];
    const int   Wl = shapes[lj * 2 + 1];
    const float px = pref[((size_t)bq * L_ + lj) * 2];
    const float py = pref[((size_t)bq * L_ + lj) * 2 + 1];
    // loc = p + off/(W,H); pixel = loc*(W,H) - 0.5  ==>  simplifies:
    const float xs = px * (float)Wl + ox - 0.5f;
    const float ys = py * (float)Hl + oy - 0.5f;

    // ---- level constants in registers (literal index after full unroll) ----
    int Harr[L_], Warr[L_], Base[L_];
#pragma unroll
    for (int l = 0; l < L_; ++l) {
        Harr[l] = shapes[l * 2];
        Warr[l] = shapes[l * 2 + 1];
        Base[l] = level_index[l];
    }

    float acc = 0.0f;
    const float* vpb = v_p + ((size_t)(b * M_ + m) * (size_t)NV_) * D_;

#pragma unroll
    for (int jj = 0; jj < 16; ++jj) {
        const int ll = jj >> 2;
        const int H = Harr[ll], W = Warr[ll], base = Base[ll];
        const float xj = lane_bcast_f(xs, jj);
        const float yj = lane_bcast_f(ys, jj);
        const float wj = lane_bcast_f(w, jj);
        const float x0f = floorf(xj), y0f = floorf(yj);
        const float lx = xj - x0f, ly = yj - y0f;
        const int x0 = (int)x0f, y0 = (int)y0f;

#pragma unroll
        for (int c = 0; c < 4; ++c) {
            const int xi = x0 + (c & 1);
            const int yi = y0 + (c >> 1);
            const float cwx = (c & 1) ? lx : (1.0f - lx);
            const float cwy = (c >> 1) ? ly : (1.0f - ly);
            const bool valid = (xi >= 0) && (xi < W) && (yi >= 0) && (yi < H);
            const int xc = min(max(xi, 0), W - 1);
            const int yc = min(max(yi, 0), H - 1);
            const size_t idx = (size_t)(base + yc * W + xc);
            const float vv = vpb[idx * D_ + lane];   // wave-uniform idx -> coalesced
            acc += wj * (valid ? (cwx * cwy) : 0.0f) * vv;
        }
    }

    tOut[(size_t)bq * C_ + m * D_ + lane] = acc;
}

// ---------------------------------------------------------------------------
extern "C" void kernel_launch(void* const* d_in, const int* in_sizes, int n_in,
                              void* d_out, int out_size, void* d_ws, size_t ws_size,
                              hipStream_t stream) {
    const float* q          = (const float*)d_in[0];
    const float* p          = (const float*)d_in[1];
    const float* v          = (const float*)d_in[2];
    const int*   shapes     = (const int*)  d_in[3];
    const int*   level_idx  = (const int*)  d_in[4];
    const float* W_off      = (const float*)d_in[5];
    const float* b_off      = (const float*)d_in[6];
    const float* W_attn     = (const float*)d_in[7];
    const float* b_attn     = (const float*)d_in[8];
    const float* W_val      = (const float*)d_in[9];
    const float* b_val      = (const float*)d_in[10];
    const float* W_out      = (const float*)d_in[11];
    const float* b_out      = (const float*)d_in[12];
    float* out = (float*)d_out;

    // ---- workspace carve (aligned 256B) ----
    char* w = (char*)d_ws;
    size_t o = 0;
    auto up = [](size_t x) { return (x + 255u) & ~(size_t)255u; };
    bf16_t* WvT = (bf16_t*)(w + o); o = up(o + (size_t)C_ * C_ * 2);
    bf16_t* WoT = (bf16_t*)(w + o); o = up(o + (size_t)C_ * C_ * 2);
    bf16_t* WaT = (bf16_t*)(w + o); o = up(o + (size_t)C_ * 128 * 2);
    bf16_t* WuT = (bf16_t*)(w + o); o = up(o + (size_t)C_ * C_ * 2);
    float* v_p      = (float*)(w + o); o = up(o + (size_t)BN_ * C_ * 4);
    float* attn_raw = (float*)(w + o); o = up(o + (size_t)BQ_ * 128 * 4);
    float* off_raw  = (float*)(w + o); o = up(o + (size_t)BQ_ * C_ * 4);
    float* tbuf     = (float*)(w + o); o = up(o + (size_t)BQ_ * C_ * 4);
    (void)ws_size; (void)in_sizes; (void)n_in; (void)out_size;

    // ---- 0: weight transpose+convert to bf16 ----
    {
        int n = C_ * C_;
        convert_wT_kernel<<<(n + 255) / 256, 256, 0, stream>>>(W_val, WvT, C_, C_);
        convert_wT_kernel<<<(n + 255) / 256, 256, 0, stream>>>(W_off, WoT, C_, C_);
        convert_wT_kernel<<<(C_ * 128 + 255) / 256, 256, 0, stream>>>(W_attn, WaT, C_, 128);
        convert_wT_kernel<<<(n + 255) / 256, 256, 0, stream>>>(W_out, WuT, C_, C_);
    }
    // ---- 1: value projection -> v_p [B][M][NV][D] ----
    {
        int waves = (BN_ / 16) * (C_ / 64);
        gemm_wmma_bf16_kernel<1><<<(waves + 7) / 8, 256, 0, stream>>>(
            v, WvT, b_val, v_p, BN_, C_, C_);
    }
    // ---- 2: attention logits [BQ][128] + offsets [BQ][256] ----
    {
        int waves = (BQ_ / 16) * (128 / 64);
        gemm_wmma_bf16_kernel<0><<<(waves + 7) / 8, 256, 0, stream>>>(
            q, WaT, b_attn, attn_raw, BQ_, C_, 128);
    }
    {
        int waves = (BQ_ / 16) * (C_ / 64);
        gemm_wmma_bf16_kernel<0><<<(waves + 7) / 8, 256, 0, stream>>>(
            q, WoT, b_off, off_raw, BQ_, C_, C_);
    }
    // ---- 3: softmax + bilinear sampling -> tbuf [BQ][C] ----
    {
        int waves = B_ * Q_ * M_;
        msda_sample_kernel<<<(waves + 7) / 8, 256, 0, stream>>>(
            attn_raw, off_raw, p, v_p, shapes, level_idx, tbuf);
    }
    // ---- 4: output projection -> d_out [BQ][C] ----
    {
        int waves = (BQ_ / 16) * (C_ / 64);
        gemm_wmma_bf16_kernel<0><<<(waves + 7) / 8, 256, 0, stream>>>(
            tbuf, WuT, b_out, out, BQ_, C_, C_);
    }
}
